// TraceableSelfAttention_66709432041879
// MI455X (gfx1250) — matrix operand order repaired and checked by reading, not compiled, not run on hardware
//
#include <hip/hip_runtime.h>

// ---------------------------------------------------------------------------
// Causal self-attention forward for MI455X (gfx1250, wave32, WMMA).
// Roofline: ~69 GFLOP vs ~100MB HBM traffic @ 23.3 TB/s => compute-bound.
// => run every matmul through v_wmma_f32_16x16x32_bf16 (f32 accumulate),
//    stage pure bf16 copies through GLOBAL_LOAD_ASYNC_TO_LDS_B128 (ASYNCcnt).
// ---------------------------------------------------------------------------

typedef __bf16 bf16;
typedef __attribute__((ext_vector_type(16))) __bf16 v16bf;
typedef __attribute__((ext_vector_type(8)))  __bf16 v8bf;
typedef __attribute__((ext_vector_type(8)))  float  v8f;

#define BATCH   2
#define T_SEQ   2048
#define D_MODEL 1024
#define NH      16
#define DHEAD   64
#define NQKV    3072   // 3*NH*DHEAD

static __device__ __forceinline__ v8f vzero8f() {
  v8f v;
#pragma unroll
  for (int i = 0; i < 8; ++i) v[i] = 0.0f;
  return v;
}

// Build a 16-element bf16 fragment from two 8-element (16B) contiguous chunks.
static __device__ __forceinline__ v16bf frag16(const bf16* p0, const bf16* p1) {
  v8bf lo = *(const v8bf*)p0;
  v8bf hi = *(const v8bf*)p1;
  v16bf r;
#pragma unroll
  for (int i = 0; i < 8; ++i) { r[i] = lo[i]; r[i + 8] = hi[i]; }
  return r;
}

static __device__ __forceinline__ v8f wmma_bf16(v16bf a, v16bf b, v8f c) {
  // (neg_a, A, neg_b, B, c_mod, C, reuse_a, reuse_b)
  return __builtin_amdgcn_wmma_f32_16x16x32_bf16(false, a, false, b, (short)0, c,
                                                 false, false);
}

// Async copy 32 bytes/lane global->LDS (two B128 beats; offset:16 biases both
// the LDS dst (VDST VGPR) and the global addr per the ISA pseudocode).
// LDS offset = low 32 bits of the generic LDS pointer (flat aperture rule).
static __device__ __forceinline__ void async_copy32(bf16* lds_dst, const bf16* gsrc) {
  unsigned lds_off = (unsigned)(unsigned long long)lds_dst;
  unsigned long long ga = (unsigned long long)gsrc;
  asm volatile("global_load_async_to_lds_b128 %0, %1, off\n\t"
               "global_load_async_to_lds_b128 %0, %1, off offset:16"
               :: "v"(lds_off), "v"(ga) : "memory");
}

static __device__ __forceinline__ void wait_asynccnt0() {
  asm volatile("s_wait_asynccnt 0x0" ::: "memory");
}

// ---------------------------------------------------------------------------
// Kernel 1: qkv = x @ Wqkv  (M=4096, N=3072, K=1024), scatter into Q/K/V
//           laid out [B, H, T, DH] in bf16.  (fp32->bf16 conversion happens
//           in the staging path, so async DMA does not apply here.)
// ---------------------------------------------------------------------------
__global__ __launch_bounds__(256) void k_gemm_qkv(
    const float* __restrict__ x, const float* __restrict__ Wqkv,
    bf16* __restrict__ Q, bf16* __restrict__ K, bf16* __restrict__ V) {
  __shared__ __align__(16) bf16 sA[128][40];   // 128 x 32, pad->40
  __shared__ __align__(16) bf16 sB[32][136];   // 32 x 128, pad->136

  const int tid  = threadIdx.x;
  const int lane = tid & 31;
  const int w    = tid >> 5;
  const int wm   = (w & 3) * 32;
  const int wn   = (w >> 2) * 64;
  const int m0   = blockIdx.y * 128;
  const int n0   = blockIdx.x * 128;

  v8f acc[2][4];
#pragma unroll
  for (int i = 0; i < 2; ++i)
#pragma unroll
    for (int j = 0; j < 4; ++j) acc[i][j] = vzero8f();

  const int ar = tid >> 1, ac = (tid & 1) * 16;
  const int br = tid >> 3, bc = (tid & 7) * 16;

  for (int k0 = 0; k0 < D_MODEL; k0 += 32) {
    const float4* pA = (const float4*)(x + (size_t)(m0 + ar) * D_MODEL + k0 + ac);
#pragma unroll
    for (int q = 0; q < 4; ++q) {
      float4 f = pA[q];
      sA[ar][ac + q * 4 + 0] = (bf16)f.x;
      sA[ar][ac + q * 4 + 1] = (bf16)f.y;
      sA[ar][ac + q * 4 + 2] = (bf16)f.z;
      sA[ar][ac + q * 4 + 3] = (bf16)f.w;
    }
    const float4* pB = (const float4*)(Wqkv + (size_t)(k0 + br) * NQKV + n0 + bc);
#pragma unroll
    for (int q = 0; q < 4; ++q) {
      float4 f = pB[q];
      sB[br][bc + q * 4 + 0] = (bf16)f.x;
      sB[br][bc + q * 4 + 1] = (bf16)f.y;
      sB[br][bc + q * 4 + 2] = (bf16)f.z;
      sB[br][bc + q * 4 + 3] = (bf16)f.w;
    }
    if (k0 + 32 < D_MODEL) {  // next K-slab -> global_prefetch_b8
      __builtin_prefetch(x + (size_t)(m0 + ar) * D_MODEL + (k0 + 32) + ac, 0, 0);
      __builtin_prefetch(Wqkv + (size_t)(k0 + 32 + br) * NQKV + n0 + bc, 0, 0);
    }
    __syncthreads();

    v16bf af[2];
#pragma unroll
    for (int mi = 0; mi < 2; ++mi) {
      const int m  = wm + mi * 16 + (lane & 15);
      const int c0 = (lane >> 4) * 8;
      af[mi] = frag16(&sA[m][c0], &sA[m][c0 + 16]);
    }
    const int kk = (lane >> 4) * 8;
#pragma unroll
    for (int ni = 0; ni < 4; ++ni) {
      const int nb = wn + ni * 16;
      v16bf bfrag;
#pragma unroll
      for (int i = 0; i < 8; ++i) { bfrag[i] = sB[kk + i][nb + (lane & 15)]; bfrag[i + 8] = sB[kk + 16 + i][nb + (lane & 15)]; }
#pragma unroll
      for (int mi = 0; mi < 2; ++mi) acc[mi][ni] = wmma_bf16(af[mi], bfrag, acc[mi][ni]);
    }
    __syncthreads();
  }

#pragma unroll
  for (int mi = 0; mi < 2; ++mi)
#pragma unroll
    for (int ni = 0; ni < 4; ++ni)
#pragma unroll
      for (int r = 0; r < 8; ++r) {
        const int m = m0 + wm + mi * 16 + ((lane >> 4) * 8) + r;
        const int n = n0 + wn + ni * 16 + (lane & 15);
        const int s   = n >> 10;
        const int rem = n & 1023;
        const int h   = rem >> 6;
        const int d   = rem & 63;
        const int bb  = m >> 11;
        const int t   = m & 2047;
        const size_t dst = (((size_t)bb * NH + h) * T_SEQ + t) * DHEAD + d;
        const bf16 val = (bf16)acc[mi][ni][r];
        if (s == 0) Q[dst] = val;
        else if (s == 1) K[dst] = val;
        else V[dst] = val;
      }
}

// ---------------------------------------------------------------------------
// Kernel 2: flash attention per (b,h).  Block = 128 thr = 4 waves; each wave
// owns a 16-row Q strip of a 64-row Q block.  K staged transposed through
// LDS; V staged once per block via async DMA (4 waves share it).
// ---------------------------------------------------------------------------
__global__ __launch_bounds__(128) void k_attn(
    const bf16* __restrict__ Q, const bf16* __restrict__ K,
    const bf16* __restrict__ V, const float* __restrict__ mask,
    bf16* __restrict__ O) {
  __shared__ __align__(16) bf16 sKt[64][40];    // K tile transposed [d][krow]
  __shared__ __align__(16) bf16 sV[32][72];     // V tile [krow][d], pad->72
  __shared__ __align__(16) bf16 sP[4][16][40];  // per-wave P bounce
  __shared__ float sM[32];

  const int tid  = threadIdx.x;
  const int lane = tid & 31;
  const int w    = tid >> 5;
  const int bh   = blockIdx.y;
  const int b    = bh >> 4;
  const int h    = bh & 15;
  const size_t base = (size_t)bh * T_SEQ * DHEAD;
  const int q0   = blockIdx.x * 64;
  const int qw   = q0 + w * 16;
  const float scale = 0.125f;  // DH^-0.5

  v16bf qf[2];
#pragma unroll
  for (int ks = 0; ks < 2; ++ks) {
    const int m  = lane & 15;
    const int c0 = ks * 32 + (lane >> 4) * 8;
    const bf16* qp = Q + base + (size_t)(qw + m) * DHEAD;
    qf[ks] = frag16(qp + c0, qp + c0 + 16);
  }

  float m_run[8], l_run[8];
#pragma unroll
  for (int r = 0; r < 8; ++r) { m_run[r] = -__builtin_inff(); l_run[r] = 0.0f; }
  v8f o[4];
#pragma unroll
  for (int j = 0; j < 4; ++j) o[j] = vzero8f();

  const int kr = tid >> 2, d0 = (tid & 3) * 16;  // staging assignment (32 rows x 4 chunks)

  for (int kt0 = 0; kt0 < q0 + 64; kt0 += 32) {
    __syncthreads();  // previous iteration's sKt/sP/sV readers are done

    // ---- async DMA: V tile (32 x 64 bf16) global -> LDS, shared by 4 waves ----
    async_copy32(&sV[kr][d0], V + base + (size_t)(kt0 + kr) * DHEAD + d0);

    // ---- stage K tile transposed into LDS: sKt[d][krow] ----
    const bf16* kp = K + base + (size_t)(kt0 + kr) * DHEAD + d0;
    v8bf k0v = *(const v8bf*)kp;
    v8bf k1v = *(const v8bf*)(kp + 8);
#pragma unroll
    for (int i = 0; i < 8; ++i) {
      sKt[d0 + i][kr]     = k0v[i];
      sKt[d0 + 8 + i][kr] = k1v[i];
    }
    if (tid < 32) sM[tid] = mask[(size_t)b * T_SEQ + kt0 + tid];
    wait_asynccnt0();   // drain this wave's async DMA before the barrier
    __syncthreads();

    // ---- S = Q @ K^T : two 16x16 tiles, K-dim 64 chained as 2x K=32 ----
    v8f s[2];
    s[0] = vzero8f();
    s[1] = vzero8f();
#pragma unroll
    for (int tn = 0; tn < 2; ++tn) {
#pragma unroll
      for (int ks = 0; ks < 2; ++ks) {
        const int dr = ks * 32 + (lane >> 4) * 8;
        v16bf bfrag;
#pragma unroll
        for (int i = 0; i < 8; ++i) { bfrag[i] = sKt[dr + i][tn * 16 + (lane & 15)]; bfrag[i + 8] = sKt[dr + 16 + i][tn * 16 + (lane & 15)]; }
        s[tn] = wmma_bf16(qf[ks], bfrag, s[tn]);
      }
    }

    // ---- scale + causal + attention mask ----
#pragma unroll
    for (int tn = 0; tn < 2; ++tn)
#pragma unroll
      for (int r = 0; r < 8; ++r) {
        const int kg = kt0 + tn * 16 + (lane & 15);
        const int qg = qw + ((lane >> 4) * 8) + r;
        float v = s[tn][r] * scale;
        if (kg > qg || sM[tn * 16 + (lane & 15)] == 0.0f) v = -__builtin_inff();
        s[tn][r] = v;
      }

    // ---- online softmax: row r lives in one 16-lane half; xor-reduce 1..8 ----
#pragma unroll
    for (int r = 0; r < 8; ++r) {
      float mx = fmaxf(s[0][r], s[1][r]);
#pragma unroll
      for (int off = 1; off < 16; off <<= 1) mx = fmaxf(mx, __shfl_xor(mx, off, 32));
      const float m_new = fmaxf(m_run[r], mx);
      const float alpha = __expf(m_run[r] - m_new);
      const float p0 = __expf(s[0][r] - m_new);
      const float p1 = __expf(s[1][r] - m_new);
      float sm = p0 + p1;
#pragma unroll
      for (int off = 1; off < 16; off <<= 1) sm += __shfl_xor(sm, off, 32);
      l_run[r] = l_run[r] * alpha + sm;
      m_run[r] = m_new;
#pragma unroll
      for (int j = 0; j < 4; ++j) o[j][r] = o[j][r] * alpha;
      const int m = ((lane >> 4) * 8) + r;
      sP[w][m][(lane & 15)]      = (bf16)p0;
      sP[w][m][16 + (lane & 15)] = (bf16)p1;
    }
    __syncthreads();  // make P visible across lanes (C layout -> A layout)

    // ---- O += P @ V : P is A (16x32), V tiles are B (32x16) from LDS ----
    v16bf pa;
    {
      const int m  = lane & 15;
      const int c0 = (lane >> 4) * 8;
      pa = frag16(&sP[w][m][c0], &sP[w][m][c0 + 16]);
    }
    const int krow = (lane >> 4) * 8;
#pragma unroll
    for (int j = 0; j < 4; ++j) {
      v16bf bfrag;
#pragma unroll
      for (int i = 0; i < 8; ++i) { bfrag[i] = sV[krow + i][j * 16 + (lane & 15)]; bfrag[i + 8] = sV[krow + 16 + i][j * 16 + (lane & 15)]; }
      o[j] = wmma_bf16(pa, bfrag, o[j]);
    }
  }

  // ---- epilogue: normalize and write O[b,t,h*64+d] ----
#pragma unroll
  for (int j = 0; j < 4; ++j)
#pragma unroll
    for (int r = 0; r < 8; ++r) {
      const int t   = qw + ((lane >> 4) * 8) + r;
      const int col = h * DHEAD + j * 16 + (lane & 15);
      const float val = o[j][r] / l_run[r];
      O[((size_t)b * T_SEQ + t) * D_MODEL + col] = (bf16)val;
    }
}

// ---------------------------------------------------------------------------
// Kernel 3: out = O @ Wo  (M=4096, N=1024, K=1024), fp32 output.
// A tile is already bf16: pure copy -> async DMA global->LDS.
// ---------------------------------------------------------------------------
__global__ __launch_bounds__(256) void k_gemm_out(
    const bf16* __restrict__ Oin, const float* __restrict__ Wo,
    float* __restrict__ out) {
  __shared__ __align__(16) bf16 sA[128][40];
  __shared__ __align__(16) bf16 sB[32][136];

  const int tid  = threadIdx.x;
  const int lane = tid & 31;
  const int w    = tid >> 5;
  const int wm   = (w & 3) * 32;
  const int wn   = (w >> 2) * 64;
  const int m0   = blockIdx.y * 128;
  const int n0   = blockIdx.x * 128;

  v8f acc[2][4];
#pragma unroll
  for (int i = 0; i < 2; ++i)
#pragma unroll
    for (int j = 0; j < 4; ++j) acc[i][j] = vzero8f();

  const int ar = tid >> 1, ac = (tid & 1) * 16;
  const int br = tid >> 3, bc = (tid & 7) * 16;

  for (int k0 = 0; k0 < D_MODEL; k0 += 32) {
    // A tile bf16: async DMA, 32B per thread (sA row pad 40 is per-lane VDST,
    // so padding is preserved; both 16B beats land via offset:16)
    async_copy32(&sA[ar][ac], Oin + (size_t)(m0 + ar) * D_MODEL + k0 + ac);

    // B tile fp32 -> bf16 through VGPRs
    const float4* pB = (const float4*)(Wo + (size_t)(k0 + br) * D_MODEL + n0 + bc);
#pragma unroll
    for (int q = 0; q < 4; ++q) {
      float4 f = pB[q];
      sB[br][bc + q * 4 + 0] = (bf16)f.x;
      sB[br][bc + q * 4 + 1] = (bf16)f.y;
      sB[br][bc + q * 4 + 2] = (bf16)f.z;
      sB[br][bc + q * 4 + 3] = (bf16)f.w;
    }
    if (k0 + 32 < D_MODEL) {
      __builtin_prefetch(Oin + (size_t)(m0 + ar) * D_MODEL + (k0 + 32) + ac, 0, 0);
      __builtin_prefetch(Wo + (size_t)(k0 + 32 + br) * D_MODEL + n0 + bc, 0, 0);
    }
    wait_asynccnt0();
    __syncthreads();

    v16bf af[2];
#pragma unroll
    for (int mi = 0; mi < 2; ++mi) {
      const int m  = wm + mi * 16 + (lane & 15);
      const int c0 = (lane >> 4) * 8;
      af[mi] = frag16(&sA[m][c0], &sA[m][c0 + 16]);
    }
    const int kk = (lane >> 4) * 8;
#pragma unroll
    for (int ni = 0; ni < 4; ++ni) {
      const int nb = wn + ni * 16;
      v16bf bfrag;
#pragma unroll
      for (int i = 0; i < 8; ++i) { bfrag[i] = sB[kk + i][nb + (lane & 15)]; bfrag[i + 8] = sB[kk + 16 + i][nb + (lane & 15)]; }
#pragma unroll
      for (int mi = 0; mi < 2; ++mi) acc[mi][ni] = wmma_bf16(af[mi], bfrag, acc[mi][ni]);
    }
    __syncthreads();
  }

#pragma unroll
  for (int mi = 0; mi < 2; ++mi)
#pragma unroll
    for (int ni = 0; ni < 4; ++ni)
#pragma unroll
      for (int r = 0; r < 8; ++r) {
        const int m = m0 + wm + mi * 16 + ((lane >> 4) * 8) + r;
        const int n = n0 + wn + ni * 16 + (lane & 15);
        out[(size_t)m * D_MODEL + n] = acc[mi][ni][r];
      }
}

// ---------------------------------------------------------------------------
// Launcher.  Workspace layout (bf16 elements):
//   Q: [B,H,T,DH] 4,194,304   K: same   V: same   O: [B,T,H*DH] 4,194,304
//   total = 32 MiB.
// ---------------------------------------------------------------------------
extern "C" void kernel_launch(void* const* d_in, const int* in_sizes, int n_in,
                              void* d_out, int out_size, void* d_ws, size_t ws_size,
                              hipStream_t stream) {
  const float* x    = (const float*)d_in[0];
  const float* Wqkv = (const float*)d_in[1];
  const float* Wo   = (const float*)d_in[2];
  const float* mask = (const float*)d_in[3];
  float* out = (float*)d_out;

  const size_t per = (size_t)BATCH * NH * T_SEQ * DHEAD;  // 4,194,304 elements
  bf16* Q = (bf16*)d_ws;
  bf16* K = Q + per;
  bf16* V = K + per;
  bf16* O = V + per;

  dim3 g1(NQKV / 128, (BATCH * T_SEQ) / 128);    // 24 x 32
  k_gemm_qkv<<<g1, 256, 0, stream>>>(x, Wqkv, Q, K, V);

  dim3 g2(T_SEQ / 64, BATCH * NH);               // 32 x 32
  k_attn<<<g2, 128, 0, stream>>>(Q, K, V, mask, O);

  dim3 g3(D_MODEL / 128, (BATCH * T_SEQ) / 128); // 8 x 32
  k_gemm_out<<<g3, 256, 0, stream>>>(O, Wo, out);
}
